// MAB_graph_16965120819752
// MI455X (gfx1250) — compile-verified, matrix operand order checked by reading, not run on hardware
//
#include <hip/hip_runtime.h>
#include <math.h>

// Problem constants (from reference)
#define B_SZ   16
#define N_KEYS 3072
#define DKI    512
#define DVO    512
#define DE     64
#define HH     8
#define LQ     21
#define DH     64          // DV / H
#define CHUNK  512         // keys per flash chunk
#define MAXCH  2           // ceil(max d (=1024) / CHUNK)
#define TK     32          // keys per inner tile (two 16-row WMMA M-tiles)
#define SLOTS  (B_SZ*HH*3*MAXCH)        // 768 partial slots
#define SCALEV 0.44194173824159216f     // 1 / (sqrt(512) * 0.1)

typedef __attribute__((ext_vector_type(16))) __bf16 v16bf;
typedef __attribute__((ext_vector_type(2)))  __bf16 v2bf;
typedef __attribute__((ext_vector_type(8)))  float  v8f;
typedef __attribute__((ext_vector_type(4)))  unsigned int u32x4;

union V16 { v16bf v; u32x4 u[2]; };

// f32 -> bf16 RNE via native fptrunc (lets the backend pick v_cvt_*bf16* on gfx1250)
__device__ __forceinline__ __bf16 f2bf1(float f) { return (__bf16)f; }

// packed f32x2 -> bf16x2; prefer hardware V_CVT_PK_BF16_F32 when clang exposes it,
// else rely on the backend pairing two fptruncs.
__device__ __forceinline__ v2bf f2bf2(float a, float b) {
#if __has_builtin(__builtin_amdgcn_cvt_pk_bf16_f32)
  return __builtin_bit_cast(v2bf, __builtin_amdgcn_cvt_pk_bf16_f32(a, b));
#else
  v2bf r; r[0] = (__bf16)a; r[1] = (__bf16)b; return r;
#endif
}

// Load one WMMA 16-bit A/B fragment per ISA layout:
// per-lane: elements 0..7  -> K = cbase + hi*8 + 0..7
//           elements 8..15 -> K = cbase + 16 + hi*8 + 0..7
// i.e. two contiguous 16-byte reads from the (row|col)-major line.
__device__ __forceinline__ v16bf load_tile16(const __bf16* rowp, int cbase, int hi) {
  V16 t;
  const __bf16* p = rowp + cbase + (hi << 3);
  t.u[0] = *(const u32x4*)(p);
  t.u[1] = *(const u32x4*)(p + 16);
  return t.v;
}

// ---------------- weight pack: f32 -> bf16, transposed (out-dim major) ----------------
__global__ void k_pack_weights(const float* WkG, const float* WkD, const float* WkL,
                               const float* WvG, const float* WvD, const float* WvL,
                               __bf16* wt) {
  int idx = blockIdx.x * blockDim.x + threadIdx.x;
  if (idx >= 6*512*512) return;
  int m = idx >> 18;
  int rem = idx & ((1 << 18) - 1);
  int o = rem >> 9;            // output dim
  int i = rem & 511;           // input channel (contiguous in wt)
  const float* src = (m==0)?WkG:(m==1)?WkD:(m==2)?WkL:(m==3)?WvG:(m==4)?WvD:WvL;
  wt[idx] = f2bf1(src[i*512 + o]);
}

// ---------------- Qp = concat(Q[:, :3]@WqG, Q[:,3:9]@WqD, Q[:,9:]@WqL) ----------------
__global__ void k_qp(const float* Q, const float* WqG, const float* WqD, const float* WqL,
                     float* Qp) {
  int idx = blockIdx.x * blockDim.x + threadIdx.x;
  if (idx >= B_SZ*LQ*DVO) return;
  int j = idx & 511;
  int q = (idx >> 9) % LQ;
  int b = idx / (LQ*DVO);
  const float* w  = (q < 3) ? WqG : (q < 9) ? WqD : WqL;
  const float* qr = Q + (size_t)(b*LQ + q)*DE;
  float s = 0.f;
  #pragma unroll 8
  for (int k = 0; k < DE; ++k) s += qr[k] * w[k*DVO + j];
  Qp[idx] = s;
}

// ---------------- wd decorrelation scalar ----------------
__global__ void k_wd(const float* Qp, float* out_wd) {
  __shared__ float red[256];
  int tid = threadIdx.x;
  float local = 0.f;
  for (int combo = tid; combo < HH*B_SZ*3; combo += 256) {
    int h = combo / (B_SZ*3);
    int rem = combo % (B_SZ*3);
    int b = rem / 3, g = rem % 3;
    int r0 = (g==0)?0:(g==1)?3:9;
    int nr = (g==0)?3:(g==1)?6:12;
    const float* base = Qp + (size_t)b*LQ*DVO + h*DH;
    float fro = 0.f;
    for (int i = 0; i < nr; ++i)
      for (int j = 0; j < nr; ++j) {
        const float* ri = base + (size_t)(r0+i)*DVO;
        const float* rj = base + (size_t)(r0+j)*DVO;
        float dot = 0.f;
        for (int k = 0; k < DH; ++k) dot += ri[k]*rj[k];
        float dd = dot - ((i==j) ? 1.f : 0.f);
        fro += dd*dd;
      }
    local += sqrtf(fro);
  }
  red[tid] = local;
  __syncthreads();
  for (int s = 128; s > 0; s >>= 1) { if (tid < s) red[tid] += red[tid+s]; __syncthreads(); }
  if (tid == 0) *out_wd = red[0] / (float)B_SZ;
}

// ---------------- fused on-the-fly K/V projection + flash attention ----------------
// grid = (MAXCH, 3 regions, B*H); block = 128 (4 wave32). Wave w owns head dims [16w,16w+16).
// 32-key tiles: two WMMA M-tiles per wave, shared B fragments, ~60 KB LDS.
__global__ void __launch_bounds__(128) k_fused_attn(
    const float* __restrict__ Kg, const float* __restrict__ Xg,
    const int* __restrict__ numv, const float* __restrict__ Qp,
    const __bf16* __restrict__ wt,
    const float* __restrict__ bvG, const float* __restrict__ bvD, const float* __restrict__ bvL,
    float* __restrict__ pm, float* __restrict__ pl, float* __restrict__ pacc)
{
  const int c = blockIdx.x, r = blockIdx.y;
  const int b = blockIdx.z / HH, h = blockIdx.z % HH;
  const int d = numv[b+1] - numv[b];
  if (c*CHUNK >= d) return;                 // whole block exits; combine kernel knows via numv
  const int cnt_total = min(CHUNK, d - c*CHUNK);
  const int start = r*d + c*CHUNK;          // region r covers keys [r*d, (r+1)*d)
  const int tiles = (cnt_total + TK - 1) / TK;

  const int tid  = threadIdx.x;
  const int lane = tid & 31, wv = tid >> 5;
  const int ln   = lane & 15, hi = lane >> 4;

  __shared__ __align__(16) __bf16 s_ab[TK*512];   // staged K rows, then origin_x rows (32 KB)
  __shared__ __align__(16) __bf16 s_q[32*64];     // Qh, zero-padded to 32 rows
  __shared__ __align__(16) __bf16 s_kh[TK*64];    // projected keys (bf16)
  __shared__ float s_vh[TK*64];                   // projected values (f32)
  __shared__ float s_S[32*TK];
  __shared__ float s_P[21*TK];
  __shared__ float s_acc[21*64];
  __shared__ float s_m[32], s_l[32], s_al[32];

  // stage Qh (zero-pad rows >= 21 so garbage never enters WMMA), init flash state
  for (int e = tid; e < 32*64; e += 128) {
    int q = e >> 6, dd = e & 63;
    float v = (q < LQ) ? Qp[(size_t)(b*LQ + q)*DVO + h*DH + dd] : 0.f;
    s_q[e] = f2bf1(v);
  }
  for (int e = tid; e < 21*64; e += 128) s_acc[e] = 0.f;
  if (tid < 32) { s_m[tid] = -INFINITY; s_l[tid] = 0.f; s_al[tid] = 0.f; }

  const __bf16* wtK = wt + (size_t)r      *512*512;
  const __bf16* wtV = wt + (size_t)(3 + r)*512*512;
  const float* bv   = (r==0) ? bvG : (r==1) ? bvD : bvL;
  const float  bias = bv[h*DH + wv*16 + ln];
  const __bf16* wkrow = wtK + (size_t)(h*DH + wv*16 + ln)*512;  // B fragment rows (L2-resident)
  const __bf16* wvrow = wtV + (size_t)(h*DH + wv*16 + ln)*512;

  const float* Kb = Kg + (size_t)b*N_KEYS*DKI;
  const float* Xb = Xg + (size_t)b*N_KEYS*DKI;

  const int qt = wv >> 1, nt = wv & 1;      // score work split: (query-tile, key-tile) per wave
  const int col = wv*16 + ln;               // this lane's head dim for projection output

  for (int t = 0; t < tiles; ++t) {
    const int k0  = start + t*TK;
    const int cnt = min(TK, cnt_total - t*TK);

    // ---- phase 1: stage K rows (f32 -> bf16), prefetch origin_x rows ----
    __syncthreads();                        // prev PV done; staging buffer free
    for (int e = tid; e < TK*128; e += 128) {
      int rr = e >> 7, c4 = (e & 127) << 2;
      float4 vk = make_float4(0.f,0.f,0.f,0.f);
      if (rr < cnt) vk = *(const float4*)(Kb + (size_t)(k0+rr)*DKI + c4);
      __bf16* o = s_ab + rr*512 + c4;
      *(v2bf*)(o)     = f2bf2(vk.x, vk.y);
      *(v2bf*)(o + 2) = f2bf2(vk.z, vk.w);
    }
    if (tid < cnt) __builtin_prefetch(Xb + (size_t)(k0 + tid)*DKI, 0, 0);
    __syncthreads();

    // ---- phase 2: kh = K_tile @ Wk_sel[:, head slice] ----
    {
      v8f ka0 = {0,0,0,0,0,0,0,0}, ka1 = {0,0,0,0,0,0,0,0};
      const __bf16* arow0 = s_ab + ln*512;
      const __bf16* arow1 = s_ab + (16 + ln)*512;
      #pragma unroll 4
      for (int ks = 0; ks < 16; ++ks) {
        int cb = ks*32;
        v16bf bK = load_tile16(wkrow, cb, hi);
        v16bf a0 = load_tile16(arow0, cb, hi);
        v16bf a1 = load_tile16(arow1, cb, hi);
        ka0 = __builtin_amdgcn_wmma_f32_16x16x32_bf16(false, a0, false, bK, (short)0, ka0, false, false);
        ka1 = __builtin_amdgcn_wmma_f32_16x16x32_bf16(false, a1, false, bK, (short)0, ka1, false, false);
      }
      int mb = hi*8;
      #pragma unroll
      for (int i = 0; i < 8; ++i) {               // D: row = i + 8*(lane>=16), col = ln
        s_kh[(mb + i)*64 + col]      = f2bf1(ka0[i]);
        s_kh[(16 + mb + i)*64 + col] = f2bf1(ka1[i]);
      }
    }
    __syncthreads();                        // s_ab reads done; s_kh visible

    // ---- phase 3: stage origin_x rows, prefetch next tile's K rows ----
    for (int e = tid; e < TK*128; e += 128) {
      int rr = e >> 7, c4 = (e & 127) << 2;
      float4 vx = make_float4(0.f,0.f,0.f,0.f);
      if (rr < cnt) vx = *(const float4*)(Xb + (size_t)(k0+rr)*DKI + c4);
      __bf16* o = s_ab + rr*512 + c4;
      *(v2bf*)(o)     = f2bf2(vx.x, vx.y);
      *(v2bf*)(o + 2) = f2bf2(vx.z, vx.w);
    }
    if (t + 1 < tiles && tid < 32 && (t+1)*TK + tid < cnt_total)
      __builtin_prefetch(Kb + (size_t)(k0 + TK + tid)*DKI, 0, 0);
    __syncthreads();

    // ---- phase 4: vh = X_tile @ Wv_sel[:, head slice] + bv ----
    {
      v8f va0 = {0,0,0,0,0,0,0,0}, va1 = {0,0,0,0,0,0,0,0};
      const __bf16* arow0 = s_ab + ln*512;
      const __bf16* arow1 = s_ab + (16 + ln)*512;
      #pragma unroll 4
      for (int ks = 0; ks < 16; ++ks) {
        int cb = ks*32;
        v16bf bV = load_tile16(wvrow, cb, hi);
        v16bf a0 = load_tile16(arow0, cb, hi);
        v16bf a1 = load_tile16(arow1, cb, hi);
        va0 = __builtin_amdgcn_wmma_f32_16x16x32_bf16(false, a0, false, bV, (short)0, va0, false, false);
        va1 = __builtin_amdgcn_wmma_f32_16x16x32_bf16(false, a1, false, bV, (short)0, va1, false, false);
      }
      int mb = hi*8;
      #pragma unroll
      for (int i = 0; i < 8; ++i) {
        s_vh[(mb + i)*64 + col]      = va0[i] + bias;
        s_vh[(16 + mb + i)*64 + col] = va1[i] + bias;
      }
    }
    __syncthreads();

    // ---- phase 5: scores S = Qh (32x64) x kh^T (64x32); one (qt,nt) combo per wave ----
    {
      v8f sa = {0,0,0,0,0,0,0,0};
      const __bf16* qrow = s_q  + (size_t)(qt*16 + ln)*64;
      const __bf16* krow = s_kh + (size_t)(nt*16 + ln)*64;  // transposed read: B[K=dd][n=key]
      #pragma unroll
      for (int ks = 0; ks < 2; ++ks) {
        v16bf aQ = load_tile16(qrow, ks*32, hi);
        v16bf bH = load_tile16(krow, ks*32, hi);
        sa = __builtin_amdgcn_wmma_f32_16x16x32_bf16(false, aQ, false, bH, (short)0, sa, false, false);
      }
      int mb = hi*8;
      #pragma unroll
      for (int i = 0; i < 8; ++i)
        s_S[(qt*16 + mb + i)*TK + nt*16 + ln] = sa[i] * SCALEV;
    }
    __syncthreads();

    // ---- phase 6: online softmax update (only k < cnt are real keys) ----
    if (wv == 0 && lane < LQ) {
      int q = lane;
      float mold = s_m[q], mx = mold;
      for (int k = 0; k < cnt; ++k) mx = fmaxf(mx, s_S[q*TK + k]);
      float alpha = __expf(mold - mx);                // exp(-inf)=0 on first tile
      float ssum = 0.f;
      for (int k = 0; k < TK; ++k) {
        float p = (k < cnt) ? __expf(s_S[q*TK + k] - mx) : 0.f;
        s_P[q*TK + k] = p; ssum += p;
      }
      s_l[q] = s_l[q]*alpha + ssum;
      s_m[q] = mx;
      s_al[q] = alpha;
    }
    __syncthreads();

    // ---- phase 7: rescale + PV accumulate; wave w owns its 16 head dims ----
    for (int e = lane; e < 21*16; e += 32) {
      int q = e >> 4, j = (e & 15) + wv*16;
      float a = s_acc[q*64 + j] * s_al[q];
      #pragma unroll 4
      for (int k = 0; k < TK; ++k) a += s_P[q*TK + k] * s_vh[k*64 + j];
      s_acc[q*64 + j] = a;
    }
  }
  __syncthreads();

  const int slot = ((b*HH + h)*3 + r)*MAXCH + c;
  if (tid < LQ) { pm[slot*32 + tid] = s_m[tid]; pl[slot*32 + tid] = s_l[tid]; }
  for (int e = tid; e < 21*64; e += 128) pacc[(size_t)slot*1344 + e] = s_acc[e];
}

// ---------------- combine flash partials across (region, chunk) ----------------
__global__ void k_combine(const int* numv, const float* pm, const float* pl,
                          const float* pacc, float* attn) {
  int b = blockIdx.x / HH, h = blockIdx.x % HH;
  int d = numv[b+1] - numv[b];
  int nc = (d + CHUNK - 1)/CHUNK;
  __shared__ float Mq[21], Lq[21];
  int tid = threadIdx.x;
  if (tid < LQ) {
    float M = -INFINITY;
    for (int r = 0; r < 3; ++r)
      for (int cc = 0; cc < nc; ++cc) {
        int slot = ((b*HH + h)*3 + r)*MAXCH + cc;
        M = fmaxf(M, pm[slot*32 + tid]);
      }
    float L = 0.f;
    for (int r = 0; r < 3; ++r)
      for (int cc = 0; cc < nc; ++cc) {
        int slot = ((b*HH + h)*3 + r)*MAXCH + cc;
        L += pl[slot*32 + tid] * __expf(pm[slot*32 + tid] - M);
      }
    Mq[tid] = M; Lq[tid] = fmaxf(L, 1e-30f);
  }
  __syncthreads();
  for (int e = tid; e < 21*64; e += 128) {
    int q = e >> 6, dd = e & 63;
    float s = 0.f;
    for (int r = 0; r < 3; ++r)
      for (int cc = 0; cc < nc; ++cc) {
        int slot = ((b*HH + h)*3 + r)*MAXCH + cc;
        s += pacc[(size_t)slot*1344 + e] * __expf(pm[slot*32 + q] - Mq[q]);
      }
    attn[(size_t)(b*LQ + q)*DVO + h*DH + dd] = s / Lq[q];
  }
}

// ---------------- epilogue: out = attn + relu(attn @ Wo + bo) ----------------
__global__ void k_mlp(const float* __restrict__ attn, const float* __restrict__ Wo,
                      const float* __restrict__ bo, float* __restrict__ out) {
  int idx = blockIdx.x * blockDim.x + threadIdx.x;
  if (idx >= B_SZ*LQ*DVO) return;
  int j = idx & 511;
  const float* row = attn + (idx & ~511);
  float acc = bo[j];
  for (int cc = 0; cc < DVO; ++cc) acc += row[cc] * Wo[cc*DVO + j];
  out[idx] = attn[idx] + fmaxf(acc, 0.f);
}

// ---------------- row L2-normalize (in place on d_out) ----------------
__global__ void k_norm(float* out) {
  __shared__ float red[256];
  float* p = out + (size_t)blockIdx.x * DVO;
  int tid = threadIdx.x;
  float ss = 0.f;
  for (int j = tid; j < DVO; j += 256) { float v = p[j]; ss += v*v; }
  red[tid] = ss; __syncthreads();
  for (int s = 128; s > 0; s >>= 1) { if (tid < s) red[tid] += red[tid+s]; __syncthreads(); }
  float inv = 1.f / fmaxf(sqrtf(red[0]), 1e-12f);
  for (int j = tid; j < DVO; j += 256) p[j] *= inv;
}

extern "C" void kernel_launch(void* const* d_in, const int* in_sizes, int n_in,
                              void* d_out, int out_size, void* d_ws, size_t ws_size,
                              hipStream_t stream) {
  const float* Q    = (const float*)d_in[0];
  const float* K    = (const float*)d_in[1];
  const float* X    = (const float*)d_in[2];
  const int*   numv = (const int*)  d_in[3];
  // d_in[4] = mask_cross: unused — mask is exactly 0 for k < 3d and -1e9 (exp underflows
  // to 0) beyond; we only visit k < 3d, which is mathematically identical.
  const float* WqG = (const float*)d_in[5];
  const float* WqD = (const float*)d_in[6];
  const float* WqL = (const float*)d_in[7];
  const float* WkG = (const float*)d_in[8];
  const float* WkD = (const float*)d_in[9];
  const float* WkL = (const float*)d_in[10];
  const float* WvG = (const float*)d_in[11];
  const float* WvD = (const float*)d_in[12];
  const float* WvL = (const float*)d_in[13];
  const float* bvG = (const float*)d_in[14];
  const float* bvD = (const float*)d_in[15];
  const float* bvL = (const float*)d_in[16];
  const float* Wo  = (const float*)d_in[17];
  const float* bo  = (const float*)d_in[18];
  float* out = (float*)d_out;

  // workspace carve-out (~8.5 MB total)
  char* ws = (char*)d_ws;
  float*  Qp  = (float*)ws;   ws += (size_t)B_SZ*LQ*DVO*4;
  __bf16* wt  = (__bf16*)ws;  ws += (size_t)6*512*512*2;
  float*  pm  = (float*)ws;   ws += (size_t)SLOTS*32*4;
  float*  pl  = (float*)ws;   ws += (size_t)SLOTS*32*4;
  float*  pacc= (float*)ws;   ws += (size_t)SLOTS*1344*4;
  float*  attn= (float*)ws;   ws += (size_t)B_SZ*LQ*DVO*4;

  k_pack_weights<<<(6*512*512 + 255)/256, 256, 0, stream>>>(WkG,WkD,WkL,WvG,WvD,WvL, wt);
  k_qp<<<(B_SZ*LQ*DVO + 255)/256, 256, 0, stream>>>(Q, WqG, WqD, WqL, Qp);
  k_wd<<<1, 256, 0, stream>>>(Qp, out + B_SZ*LQ*DVO);
  dim3 g(MAXCH, 3, B_SZ*HH);
  k_fused_attn<<<g, 128, 0, stream>>>(K, X, numv, Qp, wt, bvG, bvD, bvL, pm, pl, pacc);
  k_combine<<<B_SZ*HH, 128, 0, stream>>>(numv, pm, pl, pacc, attn);
  k_mlp<<<(B_SZ*LQ*DVO + 255)/256, 256, 0, stream>>>(attn, Wo, bo, out);
  k_norm<<<B_SZ*LQ, 256, 0, stream>>>(out);
}
